// TernaryLinear_75720273428527
// MI455X (gfx1250) — compile-verified, matrix operand order checked by reading, not run on hardware
//
#include <hip/hip_runtime.h>
#include <cstdint>
#include <cstddef>

// ---------------- CDNA5 WMMA vector types ----------------
typedef __attribute__((ext_vector_type(16))) __bf16 v16bf;
typedef __attribute__((ext_vector_type(8)))  float  v8f;

#define TOKENS 8192
#define IN_F   4096
#define OUT_F  4096

// GEMM tiling
#define KSTEP     32                    // one WMMA K per LDS tile
#define LDS_ROWB  80                    // 32 bf16 (64B) + 16B pad; 16B aligned, 20-bank row skew
#define TILE_B    (128 * LDS_ROWB)      // 10240 B per X or W panel
#define BUF_B     (2 * TILE_B)          // X + W panels per buffer
#define NKT       (IN_F / KSTEP)        // 128 k-tiles

// Round-to-nearest-even fp32 -> bf16
__device__ __forceinline__ unsigned short f32_to_bf16(float f) {
  unsigned u = __float_as_uint(f);
  u += 0x7FFFu + ((u >> 16) & 1u);
  return (unsigned short)(u >> 16);
}

// ---------------- CDNA5 async copy helpers ----------------
__device__ __forceinline__ void async_b128(uint32_t lds_addr, uint64_t gaddr) {
  // ASYNCcnt-tracked DMA of 16B/lane into LDS; vdst = LDS byte address VGPR.
  asm volatile("global_load_async_to_lds_b128 %0, %1, off"
               :: "v"(lds_addr), "v"(gaddr) : "memory");
}

__device__ __forceinline__ void wait_async0() {
#if __has_builtin(__builtin_amdgcn_s_wait_asynccnt)
  __builtin_amdgcn_s_wait_asynccnt(0);
#else
  asm volatile("s_wait_asynccnt 0x0" ::: "memory");
#endif
}

// ---------------- Kernel 1: sum(|W|) ----------------
__global__ __launch_bounds__(256) void absmean_kernel(const float* __restrict__ w,
                                                      float* __restrict__ sum) {
  __shared__ float red[256];
  const size_t N = (size_t)OUT_F * (size_t)IN_F;
  float acc = 0.f;
  for (size_t i = (size_t)blockIdx.x * 256u + threadIdx.x; i < N;
       i += (size_t)gridDim.x * 256u)
    acc += fabsf(w[i]);
  red[threadIdx.x] = acc;
  __syncthreads();
  for (int s = 128; s > 0; s >>= 1) {
    if ((int)threadIdx.x < s) red[threadIdx.x] += red[threadIdx.x + s];
    __syncthreads();
  }
  if (threadIdx.x == 0) atomicAdd(sum, red[0]);
}

// ---------------- Kernel 2: x fp32 -> bf16 ----------------
__global__ __launch_bounds__(256) void cvt_x_kernel(const float* __restrict__ x,
                                                    unsigned short* __restrict__ xb) {
  size_t i = ((size_t)blockIdx.x * 256u + threadIdx.x) * 4u;
  float4 v = *(const float4*)(x + i);
  ushort4 o;
  o.x = f32_to_bf16(v.x);
  o.y = f32_to_bf16(v.y);
  o.z = f32_to_bf16(v.z);
  o.w = f32_to_bf16(v.w);
  *(ushort4*)(xb + i) = o;
}

// ---------------- Kernel 3: ternarize W -> bf16 {-1,0,+1} ----------------
__device__ __forceinline__ unsigned short ternarize(float v, float mean) {
  float t = (fabsf(v) > mean) ? (v > 0.f ? 1.0f : -1.0f) : 0.0f;
  return f32_to_bf16(t);
}

__global__ __launch_bounds__(256) void pack_w_kernel(const float* __restrict__ w,
                                                     const float* __restrict__ sum,
                                                     unsigned short* __restrict__ wb) {
  const float mean = *sum * (1.0f / ((float)OUT_F * (float)IN_F));
  size_t i = ((size_t)blockIdx.x * 256u + threadIdx.x) * 4u;
  float4 v = *(const float4*)(w + i);
  ushort4 o;
  o.x = ternarize(v.x, mean);
  o.y = ternarize(v.y, mean);
  o.z = ternarize(v.z, mean);
  o.w = ternarize(v.w, mean);
  *(ushort4*)(wb + i) = o;
}

// ---------------- Kernel 4: WMMA bf16 GEMM, async-LDS double buffered ----------------
union FragBF { uint4 q[2]; v16bf v; };

__global__ __launch_bounds__(256) void gemm_bf16_kernel(const unsigned short* __restrict__ X,
                                                        const unsigned short* __restrict__ Wq,
                                                        float* __restrict__ C) {
  __shared__ __align__(16) unsigned char lds[2 * BUF_B];  // 40 KB

  const int tid  = threadIdx.x;
  const int lane = tid & 31;
  const int wave = tid >> 5;
  const int mblk = blockIdx.x * 128;
  const int nblk = blockIdx.y * 128;

  // ---- loader mapping: 256 threads x 16B chunks; 4 chunks/row, 64 rows/pass ----
  const int tchunk = tid & 3;   // 16B chunk within a 64B k-row
  const int trow   = tid >> 2;  // 0..63
  uint64_t gx = (uint64_t)(uintptr_t)X +
                ((uint64_t)(mblk + trow) * IN_F + (uint64_t)tchunk * 8) * 2;
  uint64_t gw = (uint64_t)(uintptr_t)Wq +
                ((uint64_t)(nblk + trow) * IN_F + (uint64_t)tchunk * 8) * 2;
  const uint64_t rowskip = (uint64_t)64 * IN_F * 2;  // +64 rows in global
  const uint32_t lds_base = (uint32_t)(uintptr_t)&lds[0];
  const uint32_t l_off = (uint32_t)trow * LDS_ROWB + (uint32_t)tchunk * 16;

  auto issue_tile = [&](int buf) {
    uint32_t xb = lds_base + (uint32_t)buf * BUF_B + l_off;
    uint32_t wb = xb + TILE_B;
    async_b128(xb,                gx);
    async_b128(xb + 64 * LDS_ROWB, gx + rowskip);
    async_b128(wb,                gw);
    async_b128(wb + 64 * LDS_ROWB, gw + rowskip);
    gx += KSTEP * 2;  // advance 64B along K
    gw += KSTEP * 2;
  };

  // ---- compute mapping: 8 waves as 2(M) x 4(N); wave tile 64x32 ----
  const int row   = lane & 15;
  const int kbyte = (lane & 16) ? 16 : 0;  // upper lanes take K+8..15 / K+24..31
  const int m0w = (wave >> 2) * 64;
  const int n0w = (wave & 3) * 32;

  uint32_t a_off[4], b_off[2];
#pragma unroll
  for (int mi = 0; mi < 4; ++mi)
    a_off[mi] = (uint32_t)(m0w + mi * 16 + row) * LDS_ROWB + kbyte;
#pragma unroll
  for (int ni = 0; ni < 2; ++ni)
    b_off[ni] = TILE_B + (uint32_t)(n0w + ni * 16 + row) * LDS_ROWB + kbyte;

  v8f acc[4][2];
  const v8f vzero = {0.f, 0.f, 0.f, 0.f, 0.f, 0.f, 0.f, 0.f};
#pragma unroll
  for (int mi = 0; mi < 4; ++mi)
#pragma unroll
    for (int ni = 0; ni < 2; ++ni)
      acc[mi][ni] = vzero;

  issue_tile(0);  // prologue

  for (int kt = 0; kt < NKT; ++kt) {
    wait_async0();     // own async loads for tile kt landed
    __syncthreads();   // tile kt visible; all waves done with buffer (kt^1)
    if (kt + 1 < NKT) issue_tile((kt + 1) & 1);

    const unsigned char* buf = lds + (size_t)(kt & 1) * BUF_B;
    v16bf a[4], b[2];
#pragma unroll
    for (int mi = 0; mi < 4; ++mi) {
      FragBF f;
      f.q[0] = *(const uint4*)(buf + a_off[mi]);
      f.q[1] = *(const uint4*)(buf + a_off[mi] + 32);
      a[mi] = f.v;
    }
#pragma unroll
    for (int ni = 0; ni < 2; ++ni) {
      FragBF f;
      f.q[0] = *(const uint4*)(buf + b_off[ni]);
      f.q[1] = *(const uint4*)(buf + b_off[ni] + 32);
      b[ni] = f.v;
    }
#pragma unroll
    for (int mi = 0; mi < 4; ++mi)
#pragma unroll
      for (int ni = 0; ni < 2; ++ni)
        acc[mi][ni] = __builtin_amdgcn_wmma_f32_16x16x32_bf16(
            false, a[mi], false, b[ni], (short)0, acc[mi][ni], false, false);
  }

  // C/D layout: VGPR i -> (M=i, N=lane) lanes 0-15; (M=i+8, N=lane-16) lanes 16-31
  const int mbase = mblk + m0w + ((lane >> 4) << 3);
  const int ncol  = nblk + n0w + row;
#pragma unroll
  for (int mi = 0; mi < 4; ++mi)
#pragma unroll
    for (int ni = 0; ni < 2; ++ni)
#pragma unroll
      for (int i = 0; i < 8; ++i)
        C[(size_t)(mbase + mi * 16 + i) * OUT_F + ncol + ni * 16] = acc[mi][ni][i];
}

// ---------------- Host launcher ----------------
extern "C" void kernel_launch(void* const* d_in, const int* in_sizes, int n_in,
                              void* d_out, int out_size, void* d_ws, size_t ws_size,
                              hipStream_t stream) {
  const float* x = (const float*)d_in[0];  // [8192, 4096] fp32
  const float* w = (const float*)d_in[1];  // [4096, 4096] fp32
  float* out = (float*)d_out;              // [8192, 4096] fp32

  const size_t xb_bytes = (size_t)TOKENS * IN_F * 2;  // 64 MiB bf16 x
  const size_t wb_bytes = (size_t)OUT_F * IN_F * 2;   // 32 MiB bf16 ternary W
  unsigned short* x_bf = (unsigned short*)d_ws;
  unsigned short* w_bf = (unsigned short*)((char*)d_ws + xb_bytes);
  float* sum = (float*)((char*)d_ws + xb_bytes + wb_bytes);

  hipMemsetAsync(sum, 0, sizeof(float), stream);

  absmean_kernel<<<1024, 256, 0, stream>>>(w, sum);
  cvt_x_kernel<<<((size_t)TOKENS * IN_F / 4) / 256, 256, 0, stream>>>(x, x_bf);
  pack_w_kernel<<<((size_t)OUT_F * IN_F / 4) / 256, 256, 0, stream>>>(w, sum, w_bf);

  dim3 grid(TOKENS / 128, OUT_F / 128);  // 64 x 32 blocks
  gemm_bf16_kernel<<<grid, 256, 0, stream>>>(x_bf, w_bf, out);
}
